// Discriminator_10651518894596
// MI455X (gfx1250) — matrix operand order repaired and checked by reading, not compiled, not run on hardware
//
#include <hip/hip_runtime.h>

// Problem constants (from reference): B=512, T=256, C=16, H=64, M=128
#define NB   512
#define NT   256
#define NC   16
#define NH   64
#define NM   128
#define NHC  1024          // H*C
#define MTILE 16           // batch rows per workgroup (WMMA M)
#define THREADS 256        // 8 wave32 waves

typedef _Float16 half_t;
typedef half_t v16h __attribute__((ext_vector_type(16)));
typedef half_t v8h  __attribute__((ext_vector_type(8)));
typedef float  v8f  __attribute__((ext_vector_type(8)));

// ---- LDS layout (bytes). Total = 280,576 (< 320 KB WGP LDS)
#define OFF_W3  0           // W3^T swizzled: 1024 x 128 f16 = 262144
#define OFF_X   262144      // X staging swizzled: 16 x 128 f16 = 4096
#define OFF_ZHH 266240      // zh f16 shadow swizzled: 16 x 64 = 2048
#define OFF_G   268288      // einsum result g: 16 x 64 f32 = 4096
#define OFF_Z   272384      // z state f32 = 4096
#define OFF_ZH  276480      // zh state f32 = 4096
#define LDS_BYTES 280576

// ---- XOR-swizzled row-major f16 tiles -------------------------------------
// Element (row,k), row length L halves, nchunks = L/8, cm = nchunks-1:
//   offset = row*L + ((chunk ^ row) & cm)*8 + (k&7),  chunk = k>>3
// 16B-granularity XOR swizzle: B-fragment loads (16 lanes reading 16 different
// rows at the same chunk) land on 16 disjoint bank quartets -> conflict-free,
// with zero padding.
__device__ __forceinline__ int swz(int row, int k, int L, int cm) {
    return row * L + ((((k >> 3) ^ row) & cm) << 3) + (k & 7);
}
__device__ __forceinline__ v8h ld8(const half_t* p, int row, int chunk, int L, int cm) {
    return *(const v8h*)(p + row * L + (((chunk ^ row) & cm) << 3));   // ds_load_b128
}
__device__ __forceinline__ v16h cat16(v8h lo, v8h hi) {
    return __builtin_shufflevector(lo, hi, 0,1,2,3,4,5,6,7,8,9,10,11,12,13,14,15);
}
// A fragment (16x32 f16 MxK): lane row m=l&15; 16 contiguous halves at
// K = k0 + g*16 (same k order as B fragments).
__device__ __forceinline__ v16h load_A(const half_t* p, int m, int k0, int g, int L, int cm) {
    return cat16(ld8(p, m, (k0 >> 3) + 2 * g,     L, cm),
                 ld8(p, m, (k0 >> 3) + 2 * g + 1, L, cm));
}
// B fragment (32x16 f16 KxN) from transposed W^T[n][k]: lane col n; 16
// contiguous halves at K = k0 + g*16 (same k order as A fragments).
__device__ __forceinline__ v16h load_B(const half_t* p, int n, int k0, int g, int L, int cm) {
    int c0 = (k0 >> 3) + 2 * g;
    return cat16(ld8(p, n, c0, L, cm), ld8(p, n, c0 + 1, L, cm));
}

// Branch-free tanh: 1 - 2/(e^{2x}+1). Saturates to +/-1, no EXEC divergence.
__device__ __forceinline__ float tanh_fast(float x) {
    float e = __expf(2.0f * x);                       // v_mul + v_exp_f32
    return 1.0f - 2.0f * __builtin_amdgcn_rcpf(e + 1.0f);
}

#define WMMA_F16(A, B, C) \
    __builtin_amdgcn_wmma_f32_16x16x32_f16(false, (A), false, (B), (short)0, (C), false, false)

// One evaluation of vf(t, zh) + fused bhc,bc->bh contraction into sG.
// Wave w owns cols [w*16,w*16+16) of GEMM1/2 and [w*128,w*128+128) of GEMM3
// (h-channels w*8..w*8+7; each 16-wide N-tile is one h with its 16 c's).
__device__ __forceinline__ void vf_eval(
    float tval,
    const half_t* sW3, half_t* sX, const half_t* sZhH, float* sG,
    int w, int ln, int g,
    const v16h rW1_0, const v16h rW1_1,
    const v16h rW2_0, const v16h rW2_1, const v16h rW2_2, const v16h rW2_3,
    float rw1r0, float rb1, float rb2, const float* rb3, const float* dy)
{
    const int colw = w * 16 + ln;

    // ---- GEMM1: X1 = tanh( zh @ W1[1:,:] + (b1 + t*W1[0,:]) ), K=64
    v8f acc = {};
    acc = WMMA_F16(load_A(sZhH, ln, 0,  g, NH, 7), rW1_0, acc);
    acc = WMMA_F16(load_A(sZhH, ln, 32, g, NH, 7), rW1_1, acc);
    {
        float bias1 = rb1 + tval * rw1r0;
#pragma unroll
        for (int i = 0; i < 8; ++i)
            sX[swz(i + 8 * g, colw, NM, 15)] = (half_t)tanh_fast(acc[i] + bias1);
    }
    __syncthreads();

    // ---- GEMM2: X2 = tanh( X1 @ W2 + b2 ), K=128 (W2 fragments in registers)
    v8f acc2 = {};
    acc2 = WMMA_F16(load_A(sX, ln, 0,  g, NM, 15), rW2_0, acc2);
    acc2 = WMMA_F16(load_A(sX, ln, 32, g, NM, 15), rW2_1, acc2);
    acc2 = WMMA_F16(load_A(sX, ln, 64, g, NM, 15), rW2_2, acc2);
    acc2 = WMMA_F16(load_A(sX, ln, 96, g, NM, 15), rW2_3, acc2);
    __syncthreads();   // all waves done READING X1 before overwrite
#pragma unroll
    for (int i = 0; i < 8; ++i)
        sX[swz(i + 8 * g, colw, NM, 15)] = (half_t)tanh_fast(acc2[i] + rb2);
    __syncthreads();

    // ---- GEMM3 (K=128, wave-local N=128) fused with einsum over C=16
    v16h a0 = load_A(sX, ln, 0,  g, NM, 15);
    v16h a1 = load_A(sX, ln, 32, g, NM, 15);
    v16h a2 = load_A(sX, ln, 64, g, NM, 15);
    v16h a3 = load_A(sX, ln, 96, g, NM, 15);
#pragma unroll
    for (int tt = 0; tt < 8; ++tt) {
        const int n = w * 128 + tt * 16 + ln;   // row of swizzled W3^T
        v8f acc3 = {};
        acc3 = WMMA_F16(a0, load_B(sW3, n, 0,  g, NM, 15), acc3);
        acc3 = WMMA_F16(a1, load_B(sW3, n, 32, g, NM, 15), acc3);
        acc3 = WMMA_F16(a2, load_B(sW3, n, 64, g, NM, 15), acc3);
        acc3 = WMMA_F16(a3, load_B(sW3, n, 96, g, NM, 15), acc3);

        const int h = w * 8 + tt;
#pragma unroll
        for (int i = 0; i < 8; ++i) {
            // row b = m0+i+8g, col c = ln: p = tanh(x3)*dy[b,c]; reduce 16 lanes (c)
            float p = tanh_fast(acc3[i] + rb3[tt]) * dy[i];
            p += __shfl_xor(p, 1);
            p += __shfl_xor(p, 2);
            p += __shfl_xor(p, 4);
            p += __shfl_xor(p, 8);
            if (ln == 0) sG[(i + 8 * g) * NH + h] = p;
        }
    }
}

__global__ void __launch_bounds__(THREADS, 1)
ncde_disc_kernel(const float* __restrict__ ys,
                 const float* __restrict__ Winit, const float* __restrict__ binit,
                 const float* __restrict__ W1, const float* __restrict__ b1,
                 const float* __restrict__ W2, const float* __restrict__ b2,
                 const float* __restrict__ W3, const float* __restrict__ b3,
                 const float* __restrict__ Wr, const float* __restrict__ br,
                 float* __restrict__ out)
{
    extern __shared__ char smem[];
    half_t* sW3  = (half_t*)(smem + OFF_W3);
    half_t* sX   = (half_t*)(smem + OFF_X);
    half_t* sZhH = (half_t*)(smem + OFF_ZHH);
    float*  sG   = (float*)(smem + OFF_G);
    float*  sZ   = (float*)(smem + OFF_Z);
    float*  sZh  = (float*)(smem + OFF_ZH);

    const int tid = threadIdx.x;
    const int w  = tid >> 5;        // wave id (wave32)
    const int l  = tid & 31;
    const int ln = l & 15;
    const int g  = l >> 4;
    const int m0 = blockIdx.x * MTILE;

    // ---------- one-time: W3 -> LDS, transposed + chunk-XOR swizzled ----------
    for (int i = tid; i < NM * NHC; i += THREADS) {     // i = k*1024 + n (coalesced in n)
        int k = i >> 10, n = i & (NHC - 1);
        sW3[swz(n, k, NM, 15)] = (half_t)W3[i];
    }

    // ---------- initial state h0 = ys[:,0,:] @ Winit + binit ----------
    for (int i = tid; i < MTILE * NH; i += THREADS) {
        int r = i >> 6, h = i & 63;
        float acc = binit[h];
#pragma unroll
        for (int c = 0; c < NC; ++c)
            acc += ys[(size_t)(m0 + r) * NT * NC + c] * Winit[c * NH + h];
        sZ[i] = acc;
        sZh[i] = acc;
        sZhH[swz(r, h, NH, 7)] = (half_t)acc;
    }

    // ---------- per-lane registers: W1/W2 B-fragments + biases ----------
    const int colw = w * 16 + ln;
    v16h rW1[2], rW2[4];
#pragma unroll
    for (int kk = 0; kk < 2; ++kk)
#pragma unroll
        for (int idx = 0; idx < 16; ++idx)
            rW1[kk][idx] = (half_t)W1[(kk * 32 + g * 16 + idx + 1) * NM + colw]; // skip time row
#pragma unroll
    for (int kk = 0; kk < 4; ++kk)
#pragma unroll
        for (int idx = 0; idx < 16; ++idx)
            rW2[kk][idx] = (half_t)W2[(kk * 32 + g * 16 + idx) * NM + colw];

    const float rw1r0 = W1[colw];     // W1 row 0 (time channel)
    const float rb1   = b1[colw];
    const float rb2   = b2[colw];
    float rb3[8];
#pragma unroll
    for (int tt = 0; tt < 8; ++tt) rb3[tt] = b3[w * 128 + tt * 16 + ln];

    __syncthreads();

    // ---------- reversible-Heun scan: 255 sequential steps ----------
    for (int t = 0; t < NT - 1; ++t) {
        float dy[8];

        // dY at t_n (c = ln, rows i+8g)
#pragma unroll
        for (int i = 0; i < 8; ++i) {
            size_t base = (size_t)(m0 + i + 8 * g) * NT * NC + (size_t)t * NC + ln;
            dy[i] = ys[base + NC] - ys[base];
            __builtin_prefetch(ys + base + 2 * NC, 0, 0);   // global_prefetch_b8
        }

        // g0 = vf(t, zh) . dY_n
        vf_eval((float)t, sW3, sX, sZhH, sG, w, ln, g,
                rW1[0], rW1[1], rW2[0], rW2[1], rW2[2], rW2[3],
                rw1r0, rb1, rb2, rb3, dy);
        __syncthreads();

        // zh <- 2z - zh + g0 ; z <- z + 0.5*g0 ; refresh f16 shadow
        for (int i = tid; i < MTILE * NH; i += THREADS) {
            int r = i >> 6, h = i & 63;
            float zo = sZ[i], zho = sZh[i], g0 = sG[i];
            float zhn = 2.0f * zo - zho + g0;
            sZh[i] = zhn;
            sZ[i]  = zo + 0.5f * g0;
            sZhH[swz(r, h, NH, 7)] = (half_t)zhn;
        }
        __syncthreads();

        // dY at t_{n+1} (clamped to last interval)
        const int t2 = (t + 1 < NT - 1) ? (t + 1) : (NT - 2);
#pragma unroll
        for (int i = 0; i < 8; ++i) {
            size_t base = (size_t)(m0 + i + 8 * g) * NT * NC + (size_t)t2 * NC + ln;
            dy[i] = ys[base + NC] - ys[base];
        }

        // g1 = vf(t+1, zh_new) . dY_{n+1}
        vf_eval((float)(t + 1), sW3, sX, sZhH, sG, w, ln, g,
                rW1[0], rW1[1], rW2[0], rW2[1], rW2[2], rW2[3],
                rw1r0, rb1, rb2, rb3, dy);
        __syncthreads();

        // z <- z + 0.5*g1
        for (int i = tid; i < MTILE * NH; i += THREADS) sZ[i] += 0.5f * sG[i];
        __syncthreads();
    }

    // ---------- readout: out = zT @ Wr + br ----------
    if (tid < MTILE) {
        float acc = br[0];
#pragma unroll 8
        for (int h = 0; h < NH; ++h) acc += sZ[tid * NH + h] * Wr[h];
        out[m0 + tid] = acc;
    }
}

extern "C" void kernel_launch(void* const* d_in, const int* in_sizes, int n_in,
                              void* d_out, int out_size, void* d_ws, size_t ws_size,
                              hipStream_t stream)
{
    const float* ys    = (const float*)d_in[0];
    const float* Winit = (const float*)d_in[1];
    const float* binit = (const float*)d_in[2];
    const float* W1    = (const float*)d_in[3];
    const float* b1    = (const float*)d_in[4];
    const float* W2    = (const float*)d_in[5];
    const float* b2    = (const float*)d_in[6];
    const float* W3    = (const float*)d_in[7];
    const float* b3    = (const float*)d_in[8];
    const float* Wr    = (const float*)d_in[9];
    const float* br    = (const float*)d_in[10];
    float* out = (float*)d_out;

    ncde_disc_kernel<<<dim3(NB / MTILE), dim3(THREADS), LDS_BYTES, stream>>>(
        ys, Winit, binit, W1, b1, W2, b2, W3, b3, Wr, br, out);
}
